// CrossAttention_30734785970426
// MI455X (gfx1250) — matrix operand order repaired and checked by reading, not compiled, not run on hardware
//
#include <hip/hip_runtime.h>

typedef __attribute__((ext_vector_type(16))) _Float16 v16h;
typedef __attribute__((ext_vector_type(8)))  _Float16 v8h;
typedef __attribute__((ext_vector_type(8)))  float    v8f;

#define DEV static __device__ __forceinline__

DEV v16h cat16(v8h lo, v8h hi) {
  v16h r;
#pragma unroll
  for (int i = 0; i < 8; ++i) { r[i] = lo[i]; r[i + 8] = hi[i]; }
  return r;
}

DEV v8f wmma16(v16h a, v16h b, v8f c) {
  // D = A(16x32,f16) * B(32x16,f16) + C(16x16,f32)
  return __builtin_amdgcn_wmma_f32_16x16x32_f16(
      /*neg_a=*/false, a, /*neg_b=*/false, b,
      /*c_mod=*/(short)0, c, /*reuse_a=*/false, /*reuse_b=*/false);
}

// A fragment: 16x32 tile from row-major matrix, src points at (row0,k0), lead dim lda.
// ISA layout: lanes 0-15 row M=lane, K={0..7}+{16..23}; lanes 16-31 K={8..15}+{24..31}.
DEV v16h load_a16x32(const _Float16* __restrict__ src, int lda) {
  int lane = threadIdx.x & 31;
  int row  = lane & 15;
  int kb   = (lane >> 4) << 3;               // 0 or 8
  const _Float16* p = src + (size_t)row * lda + kb;
  v8h lo = *(const v8h*)(p);                 // K = kb..kb+7
  v8h hi = *(const v8h*)(p + 16);            // K = 16+kb..16+kb+7
  return cat16(lo, hi);
}

// B fragment: 32x16 tile where element B[k][n] = src[n*ld + k] (i.e. src is the
// K-contiguous, N-major storage of B). Lanes 0-15: col N=lane, K={0..7}+{16..23};
// lanes 16-31: K={8..15}+{24..31}. Works for W^T from row-major W[N,K], for K^T tiles,
// and for LDS-staged tiles.
DEV v16h load_bT32x16(const _Float16* __restrict__ src, int ld) {
  int lane = threadIdx.x & 31;
  int col  = lane & 15;
  int kb   = (lane >> 4) << 3;               // 0 or 8
  const _Float16* p = src + (size_t)col * ld + kb;
  v8h lo = *(const v8h*)(p);
  v8h hi = *(const v8h*)(p + 16);
  return cat16(lo, hi);
}

// ---------------------------------------------------------------- weight cast
__global__ void cast_f16_kernel(const float* __restrict__ in,
                                _Float16* __restrict__ out, int n) {
  int i = blockIdx.x * blockDim.x + threadIdx.x;
  if (i < n) out[i] = (_Float16)in[i];
}

// ---------------------------------------------------------------- rmsnorm+cast
__global__ __launch_bounds__(256)
void rmsnorm_cast_kernel(const float* __restrict__ x, const float* __restrict__ g,
                         _Float16* __restrict__ out, int D) {
  int row = blockIdx.x;
  const float* xr = x + (size_t)row * D;
  _Float16* orow  = out + (size_t)row * D;
  float s = 0.f;
  for (int i = threadIdx.x; i < D; i += blockDim.x) { float v = xr[i]; s += v * v; }
#pragma unroll
  for (int off = 16; off > 0; off >>= 1) s += __shfl_xor(s, off);
  __shared__ float wpart[8];
  int wid = threadIdx.x >> 5, lane = threadIdx.x & 31;
  if (lane == 0) wpart[wid] = s;
  __syncthreads();
  float tot = 0.f;
#pragma unroll
  for (int i = 0; i < 8; ++i) tot += wpart[i];
  float rinv = rsqrtf(tot / (float)D + 1e-5f);
  for (int i = threadIdx.x; i < D; i += blockDim.x)
    orow[i] = (_Float16)(xr[i] * rinv * g[i]);
}

// ---------------------------------------------------------------- GEMM C=A*W^T (f16 out)
__global__ __launch_bounds__(256)
void gemm_awt_f16_kernel(const _Float16* __restrict__ A, const _Float16* __restrict__ W,
                         _Float16* __restrict__ C, int M, int N, int K) {
  int wid    = blockIdx.x * (blockDim.x >> 5) + (threadIdx.x >> 5);
  int ntiles = N >> 6;
  int total  = (M >> 4) * ntiles;
  if (wid >= total) return;                // wave-uniform
  int mt = wid / ntiles, nt = wid % ntiles;
  const _Float16* Ab = A + (size_t)mt * 16 * K;
  v8f acc[4] = {};
  for (int k0 = 0; k0 < K; k0 += 32) {
    __builtin_prefetch(Ab + k0 + 128, 0, 3);
    v16h a = load_a16x32(Ab + k0, K);
#pragma unroll
    for (int j = 0; j < 4; ++j) {
      v16h b = load_bT32x16(W + (size_t)(nt * 64 + j * 16) * K + k0, K);
      acc[j] = wmma16(a, b, acc[j]);
    }
  }
  int lane = threadIdx.x & 31;
  int col = lane & 15, rb = (lane >> 4) << 3;
#pragma unroll
  for (int j = 0; j < 4; ++j)
#pragma unroll
    for (int r = 0; r < 8; ++r) {
      size_t row = (size_t)mt * 16 + rb + r;
      C[row * N + nt * 64 + j * 16 + col] = (_Float16)acc[j][r];
    }
}

// ------------------------------------------- GEMM C^T = (A*W^T)^T  (f16, transposed store)
// Fixed shape: rows are [b*2048 + t]; output CT[(bb*N + cg)*2048 + tt].
// Used for the V projection so attention can read V with K(t)-contiguous fragments.
__global__ __launch_bounds__(256)
void gemm_awt_f16T_kernel(const _Float16* __restrict__ A, const _Float16* __restrict__ W,
                          _Float16* __restrict__ CT, int M, int N, int K) {
  constexpr int T = 2048;                  // compile-time: div/mod -> shift/mask
  int wid    = blockIdx.x * (blockDim.x >> 5) + (threadIdx.x >> 5);
  int ntiles = N >> 6;
  int total  = (M >> 4) * ntiles;
  if (wid >= total) return;
  int mt = wid / ntiles, nt = wid % ntiles;
  const _Float16* Ab = A + (size_t)mt * 16 * K;
  v8f acc[4] = {};
  for (int k0 = 0; k0 < K; k0 += 32) {
    __builtin_prefetch(Ab + k0 + 128, 0, 3);
    v16h a = load_a16x32(Ab + k0, K);
#pragma unroll
    for (int j = 0; j < 4; ++j) {
      v16h b = load_bT32x16(W + (size_t)(nt * 64 + j * 16) * K + k0, K);
      acc[j] = wmma16(a, b, acc[j]);
    }
  }
  int lane = threadIdx.x & 31;
  int col = lane & 15, rb = (lane >> 4) << 3;
  int bb = (mt * 16) >> 11;                // 16-row tile never crosses T boundary
#pragma unroll
  for (int j = 0; j < 4; ++j)
#pragma unroll
    for (int r = 0; r < 8; ++r) {
      int row = mt * 16 + rb + r;
      int tt  = row & (T - 1);
      int cg  = nt * 64 + j * 16 + col;
      CT[((size_t)bb * N + cg) * T + tt] = (_Float16)acc[j][r];
    }
}

// ---------------------------------------------------------------- GEMM C=resid+A*W^T (f32 out)
__global__ __launch_bounds__(256)
void gemm_awt_res_f32_kernel(const _Float16* __restrict__ A, const _Float16* __restrict__ W,
                             const float* __restrict__ resid, float* __restrict__ C,
                             int M, int N, int K) {
  int wid    = blockIdx.x * (blockDim.x >> 5) + (threadIdx.x >> 5);
  int ntiles = N >> 6;
  int total  = (M >> 4) * ntiles;
  if (wid >= total) return;
  int mt = wid / ntiles, nt = wid % ntiles;
  const _Float16* Ab = A + (size_t)mt * 16 * K;
  v8f acc[4] = {};
  for (int k0 = 0; k0 < K; k0 += 32) {
    __builtin_prefetch(Ab + k0 + 128, 0, 3);
    v16h a = load_a16x32(Ab + k0, K);
#pragma unroll
    for (int j = 0; j < 4; ++j) {
      v16h b = load_bT32x16(W + (size_t)(nt * 64 + j * 16) * K + k0, K);
      acc[j] = wmma16(a, b, acc[j]);
    }
  }
  int lane = threadIdx.x & 31;
  int col = lane & 15, rb = (lane >> 4) << 3;
#pragma unroll
  for (int j = 0; j < 4; ++j)
#pragma unroll
    for (int r = 0; r < 8; ++r) {
      size_t row = (size_t)mt * 16 + rb + r;
      size_t idx = row * N + nt * 64 + j * 16 + col;
      C[idx] = resid[idx] + acc[j][r];
    }
}

// ---------------------------------------------------------------- fused GQA flash attention
// Block = 8 waves, all sharing one (b, g) KV-head. Block stages each 32-key
// K tile (32x64 f16) and V^T tile (64x32 f16) into LDS with async-to-LDS DMA
// (double-buffered, ASYNCcnt). Each wave owns 16 query rows of one (r, s-tile).
__global__ __launch_bounds__(256)
void attn_kernel(const _Float16* __restrict__ Q, const _Float16* __restrict__ Km,
                 const _Float16* __restrict__ Vt, _Float16* __restrict__ O) {
  const int Sd = 2048, Td = 2048, Rr = 4, QLD = 1024, KLD = 256, LDSW = 40;

  int lane = threadIdx.x & 31;
  int lw   = threadIdx.x >> 5;              // wave in block (0..7)
  int bg    = blockIdx.x >> 6;              // (b,g): 0..7
  int chunk = blockIdx.x & 63;              // 64 chunks of 8 m-tiles
  int b = bg >> 2, g = bg & 3;
  int mt = chunk * 8 + lw;                  // 0..511 over (r, s-tile)
  int r  = mt >> 7;                         // head within group
  int st = mt & 127;                        // 16-row query tile

  __shared__ _Float16 ldsK[2][32 * 64];     // keys t0..t0+31, 64 hd each
  __shared__ _Float16 ldsV[2][64 * 32];     // V^T: hd rows, 32 t cols
  __shared__ _Float16 plds[8][16 * LDSW];   // per-wave P staging

  const _Float16* qb     = Q  + ((size_t)(b * Sd + st * 16)) * QLD + (g * Rr + r) * 64;
  const _Float16* kbase  = Km + (size_t)b * Td * KLD + g * 64;
  const _Float16* vtbase = Vt + ((size_t)b * 256 + g * 64) * Td;

  v16h qa0 = load_a16x32(qb, QLD);          // HD 0..31
  v16h qa1 = load_a16x32(qb + 32, QLD);     // HD 32..63

  v8f o0 = {}, o1 = {}, o2 = {}, o3 = {};
  float m[8], l[8];
#pragma unroll
  for (int i = 0; i < 8; ++i) { m[i] = -1e30f; l[i] = 0.f; }
  const float scale = 0.125f;               // 1/sqrt(64)

  // async staging: each wave moves 512B of K (4 rows) + 512B of V^T (8 rows)
  auto stage = [&](int buf, int t0) {
    int krow = lw * 4 + (lane >> 3);        // 0..31
    unsigned kLds = (unsigned)(uintptr_t)&ldsK[buf][krow * 64 + (lane & 7) * 8];
    unsigned long long kGlb =
        (unsigned long long)(uintptr_t)(kbase + (size_t)(t0 + krow) * KLD + (lane & 7) * 8);
    asm volatile("global_load_async_to_lds_b128 %0, %1, off"
                 :: "v"(kLds), "v"(kGlb) : "memory");
    int vrow = lw * 8 + (lane >> 2);        // 0..63 (hd)
    unsigned vLds = (unsigned)(uintptr_t)&ldsV[buf][vrow * 32 + (lane & 3) * 8];
    unsigned long long vGlb =
        (unsigned long long)(uintptr_t)(vtbase + (size_t)vrow * Td + t0 + (lane & 3) * 8);
    asm volatile("global_load_async_to_lds_b128 %0, %1, off"
                 :: "v"(vLds), "v"(vGlb) : "memory");
  };

  stage(0, 0);                              // prologue: tile 0 in flight

  for (int t0 = 0; t0 < Td; t0 += 32) {
    int buf = (t0 >> 5) & 1;
    __syncthreads();                        // prev tile's readers done -> safe to overwrite
    if (t0 + 32 < Td) {
      stage(buf ^ 1, t0 + 32);              // prefetch next tile
      asm volatile("s_wait_asynccnt 0x2" ::: "memory");  // current tile's 2 ops done
    } else {
      asm volatile("s_wait_asynccnt 0x0" ::: "memory");
    }
    __syncthreads();                        // all waves' staged data visible

    // scores: two 16x16 key tiles, K-dim = HD = 64 (two WMMAs each)
    const _Float16* kt = &ldsK[buf][0];
    v8f s0 = {}, s1 = {};
    s0 = wmma16(qa0, load_bT32x16(kt,            64), s0);
    s0 = wmma16(qa1, load_bT32x16(kt + 32,       64), s0);
    s1 = wmma16(qa0, load_bT32x16(kt + 16 * 64,      64), s1);
    s1 = wmma16(qa1, load_bT32x16(kt + 16 * 64 + 32, 64), s1);

    // online softmax; row reductions across 16-lane column groups
#pragma unroll
    for (int i = 0; i < 8; ++i) {
      float a0 = s0[i] * scale, a1 = s1[i] * scale;
      float mx = fmaxf(a0, a1);
      mx = fmaxf(mx, __shfl_xor(mx, 1));
      mx = fmaxf(mx, __shfl_xor(mx, 2));
      mx = fmaxf(mx, __shfl_xor(mx, 4));
      mx = fmaxf(mx, __shfl_xor(mx, 8));
      float mn    = fmaxf(m[i], mx);
      float alpha = __expf(m[i] - mn);
      float p0 = __expf(a0 - mn), p1 = __expf(a1 - mn);
      float rs = p0 + p1;
      rs += __shfl_xor(rs, 1); rs += __shfl_xor(rs, 2);
      rs += __shfl_xor(rs, 4); rs += __shfl_xor(rs, 8);
      l[i] = l[i] * alpha + rs;
      m[i] = mn;
      o0[i] *= alpha; o1[i] *= alpha; o2[i] *= alpha; o3[i] *= alpha;
      s0[i] = p0; s1[i] = p1;
    }

    // P (C-layout) -> LDS -> A-fragment (16x32)
    _Float16* my = plds[lw];
    {
      int col = lane & 15, rb8 = (lane >> 4) << 3;
#pragma unroll
      for (int i = 0; i < 8; ++i) {
        my[(rb8 + i) * LDSW + col]      = (_Float16)s0[i];
        my[(rb8 + i) * LDSW + 16 + col] = (_Float16)s1[i];
      }
    }
    __builtin_amdgcn_wave_barrier();
    v16h pa;
    {
      int prow = lane & 15, kb8 = (lane >> 4) << 3;
#pragma unroll
      for (int i = 0; i < 8; ++i) {
        pa[i]     = my[prow * LDSW + kb8 + i];
        pa[8 + i] = my[prow * LDSW + 16 + kb8 + i];
      }
    }
    __builtin_amdgcn_wave_barrier();

    // O += P(16x32) * V(32x64); V^T staged in LDS, K(t)-contiguous
    const _Float16* vt = &ldsV[buf][0];
    o0 = wmma16(pa, load_bT32x16(vt,           32), o0);
    o1 = wmma16(pa, load_bT32x16(vt + 16 * 32, 32), o1);
    o2 = wmma16(pa, load_bT32x16(vt + 32 * 32, 32), o2);
    o3 = wmma16(pa, load_bT32x16(vt + 48 * 32, 32), o3);
  }

  // epilogue: normalize and store f16 attention output [B*S, 1024]
  int col = lane & 15, rb8 = (lane >> 4) << 3;
#pragma unroll
  for (int i = 0; i < 8; ++i) {
    float inv = 1.f / l[i];
    size_t row = (size_t)(b * Sd + st * 16 + rb8 + i);
    _Float16* op = O + row * 1024 + (g * Rr + r) * 64;
    op[col]      = (_Float16)(o0[i] * inv);
    op[16 + col] = (_Float16)(o1[i] * inv);
    op[32 + col] = (_Float16)(o2[i] * inv);
    op[48 + col] = (_Float16)(o3[i] * inv);
  }
}

// ---------------------------------------------------------------- launch
extern "C" void kernel_launch(void* const* d_in, const int* in_sizes, int n_in,
                              void* d_out, int out_size, void* d_ws, size_t ws_size,
                              hipStream_t stream) {
  (void)in_sizes; (void)n_in; (void)out_size; (void)ws_size;
  const float* x   = (const float*)d_in[0];
  const float* kv  = (const float*)d_in[1];
  const float* wq  = (const float*)d_in[2];
  const float* wk  = (const float*)d_in[3];
  const float* wv  = (const float*)d_in[4];
  const float* wo  = (const float*)d_in[5];
  const float* gq  = (const float*)d_in[6];
  const float* gkv = (const float*)d_in[7];
  float* out = (float*)d_out;

  const int D = 1024, HHD = 1024, KVD = 256;
  const int MQ = 4096;   // B*S
  const int MK = 4096;   // B*T

  char* ws = (char*)d_ws;
  size_t off = 0;
  auto alloc = [&](size_t bytes) {
    char* p = ws + off; off += (bytes + 255) & ~(size_t)255; return p;
  };
  _Float16* xn  = (_Float16*)alloc((size_t)MQ * D * 2);
  _Float16* kvn = (_Float16*)alloc((size_t)MK * D * 2);
  _Float16* qh  = (_Float16*)alloc((size_t)MQ * HHD * 2);
  _Float16* kh  = (_Float16*)alloc((size_t)MK * KVD * 2);
  _Float16* vth = (_Float16*)alloc((size_t)MK * KVD * 2);  // V^T: [B][256][T]
  _Float16* ah  = (_Float16*)alloc((size_t)MQ * HHD * 2);
  _Float16* wqh = (_Float16*)alloc((size_t)HHD * D * 2);
  _Float16* wkh = (_Float16*)alloc((size_t)KVD * D * 2);
  _Float16* wvh = (_Float16*)alloc((size_t)KVD * D * 2);
  _Float16* woh = (_Float16*)alloc((size_t)D * HHD * 2);

  auto cdiv = [](int a, int b) { return (a + b - 1) / b; };
  cast_f16_kernel<<<cdiv(HHD * D, 256), 256, 0, stream>>>(wq, wqh, HHD * D);
  cast_f16_kernel<<<cdiv(KVD * D, 256), 256, 0, stream>>>(wk, wkh, KVD * D);
  cast_f16_kernel<<<cdiv(KVD * D, 256), 256, 0, stream>>>(wv, wvh, KVD * D);
  cast_f16_kernel<<<cdiv(D * HHD, 256), 256, 0, stream>>>(wo, woh, D * HHD);

  rmsnorm_cast_kernel<<<MQ, 256, 0, stream>>>(x, gq, xn, D);
  rmsnorm_cast_kernel<<<MK, 256, 0, stream>>>(kv, gkv, kvn, D);

  { int waves = (MQ / 16) * (HHD / 64);
    gemm_awt_f16_kernel<<<waves / 8, 256, 0, stream>>>(xn, wqh, qh, MQ, HHD, D); }
  { int waves = (MK / 16) * (KVD / 64);
    gemm_awt_f16_kernel<<<waves / 8, 256, 0, stream>>>(kvn, wkh, kh, MK, KVD, D); }
  { int waves = (MK / 16) * (KVD / 64);
    gemm_awt_f16T_kernel<<<waves / 8, 256, 0, stream>>>(kvn, wvh, vth, MK, KVD, D); }

  attn_kernel<<<512, 256, 0, stream>>>(qh, kh, vth, ah);

  { int waves = (MQ / 16) * (D / 64);
    gemm_awt_res_f32_kernel<<<waves / 8, 256, 0, stream>>>(ah, woh, x, out, MQ, D, HHD); }
}
